// QuantumKernelMethod_65481071408908
// MI455X (gfx1250) — matrix operand order repaired and checked by reading, not compiled, not run on hardware
//
#include <hip/hip_runtime.h>

typedef __attribute__((ext_vector_type(16))) _Float16 v16h;
typedef __attribute__((ext_vector_type(8)))  _Float16 v8h;
typedef __attribute__((ext_vector_type(8)))  float    v8f;
typedef __attribute__((ext_vector_type(2)))  _Float16 h2;

constexpr int N = 8192;   // rows of x
constexpr int M = 8192;   // rows of y
constexpr int D = 64;     // feature dim

// ---------------------------------------------------------------------------
// Pre-pass: convert f32 -> f16 and compute per-row squared norms.
// One wave32 per row: each lane handles 2 elements, butterfly-reduce sum of
// squares, lane 0 writes the norm. Writes are fully coalesced.
// Rows [0,N) come from x, rows [N, N+M) from y.
// ---------------------------------------------------------------------------
__global__ __launch_bounds__(256) void rbf_prep(
    const float* __restrict__ x, const float* __restrict__ y,
    _Float16* __restrict__ xh, _Float16* __restrict__ yh,
    float* __restrict__ x2, float* __restrict__ y2) {
  const int gid  = blockIdx.x * blockDim.x + threadIdx.x;
  const int lane = gid & 31;
  const int row  = gid >> 5;                 // 0 .. N+M-1

  const float*  src;  _Float16* dsth;  float* dstn;  int r;
  if (row < N) { src = x; dsth = xh; dstn = x2; r = row; }
  else         { src = y; dsth = yh; dstn = y2; r = row - N; }

  const float2 v = *(const float2*)(src + (size_t)r * D + lane * 2);
  float ss = v.x * v.x + v.y * v.y;
  #pragma unroll
  for (int off = 16; off > 0; off >>= 1) ss += __shfl_xor(ss, off, 32);

  h2 h; h.x = (_Float16)v.x; h.y = (_Float16)v.y;
  *(h2*)(dsth + (size_t)r * D + lane * 2) = h;
  if (lane == 0) dstn[r] = ss;
}

// ---------------------------------------------------------------------------
// Main kernel: 128x128 output tile per 256-thread block (8 wave32).
// Wave w: rows [ (w&3)*32, +32 ), cols [ (w>>2)*64, +64 )  => 2x4 C-tiles.
// K=64 fully unrolled as 2 WMMA k-steps of 32.
// ---------------------------------------------------------------------------
__global__ __launch_bounds__(256) void rbf_wmma(
    const _Float16* __restrict__ xh, const _Float16* __restrict__ yh,
    const float* __restrict__ x2, const float* __restrict__ y2,
    const float* __restrict__ gptr, float* __restrict__ out) {
  const float gamma = gptr[0];

  const int lane   = threadIdx.x & 31;
  const int wave   = threadIdx.x >> 5;
  const int l16    = lane & 15;
  const int half16 = lane >> 4;              // 0: lanes 0-15, 1: lanes 16-31

  const int waveRow0 = blockIdx.x * 128 + (wave & 3) * 32;   // x rows
  const int waveCol0 = blockIdx.y * 128 + (wave >> 2) * 64;  // y rows

  // ---- A fragments (x): 2 row-tiles x 2 k-steps -------------------------
  // Lane layout (16-bit A 16x32): lane<16 -> K {kb..kb+7, kb+16..kb+23},
  // kb = half16*8 within the k-step.
  v16h a[2][2];
  #pragma unroll
  for (int rt = 0; rt < 2; ++rt) {
    const _Float16* rowp = xh + (size_t)(waveRow0 + rt * 16 + l16) * D;
    #pragma unroll
    for (int ks = 0; ks < 2; ++ks) {
      const _Float16* p = rowp + ks * 32 + half16 * 8;
      v8h lo = *(const v8h*)(p);
      v8h hi = *(const v8h*)(p + 16);
      a[rt][ks] = __builtin_shufflevector(lo, hi, 0,1,2,3,4,5,6,7,
                                                  8,9,10,11,12,13,14,15);
    }
  }

  // ---- B fragments (y): 4 col-tiles x 2 k-steps -------------------------
  // Lane layout (16-bit B 32x16): lane = column, K {kb..kb+7, kb+16..kb+23},
  // kb = half16*8 within the k-step. Column n of B == contiguous row n of y.
  v16h b[4][2];
  #pragma unroll
  for (int ct = 0; ct < 4; ++ct) {
    const _Float16* rowp = yh + (size_t)(waveCol0 + ct * 16 + l16) * D;
    #pragma unroll
    for (int ks = 0; ks < 2; ++ks) {
      const _Float16* p = rowp + ks * 32 + half16 * 8;
      v8h lo = *(const v8h*)(p);
      v8h hi = *(const v8h*)(p + 16);
      b[ct][ks] = __builtin_shufflevector(lo, hi, 0,1,2,3,4,5,6,7,
                                                  8,9,10,11,12,13,14,15);
    }
  }

  // ---- Row / col norms ---------------------------------------------------
  // x2f[rt]: lane l16 holds norm of tile-row l16 (shuffled in epilogue).
  float x2f[2];
  #pragma unroll
  for (int rt = 0; rt < 2; ++rt) x2f[rt] = x2[waveRow0 + rt * 16 + l16];
  float y2f[4];
  #pragma unroll
  for (int ct = 0; ct < 4; ++ct) y2f[ct] = y2[waveCol0 + ct * 16 + l16];

  // ---- WMMA + fused epilogue --------------------------------------------
  #pragma unroll
  for (int rt = 0; rt < 2; ++rt) {
    #pragma unroll
    for (int ct = 0; ct < 4; ++ct) {
      v8f c = {};
      c = __builtin_amdgcn_wmma_f32_16x16x32_f16(false, a[rt][0],
                                                 false, b[ct][0],
                                                 (short)0, c, false, false);
      c = __builtin_amdgcn_wmma_f32_16x16x32_f16(false, a[rt][1],
                                                 false, b[ct][1],
                                                 (short)0, c, false, false);
      // C layout: VGPR r, lane L -> row r + 8*(L>>4), col L&15.
      const float y2v = y2f[ct];
      const int   col = waveCol0 + ct * 16 + l16;
      #pragma unroll
      for (int r = 0; r < 8; ++r) {
        const int rowInTile = r + 8 * half16;
        const float x2v = __shfl(x2f[rt], rowInTile, 32);
        float sq = fmaxf(x2v + y2v - 2.0f * c[r], 0.0f);
        const float e = __expf(-gamma * sq);
        const size_t row = (size_t)(waveRow0 + rt * 16 + rowInTile);
        __builtin_nontemporal_store(e, out + row * (size_t)M + col);
      }
    }
  }
}

// ---------------------------------------------------------------------------
extern "C" void kernel_launch(void* const* d_in, const int* in_sizes, int n_in,
                              void* d_out, int out_size, void* d_ws, size_t ws_size,
                              hipStream_t stream) {
  const float* x = (const float*)d_in[0];
  const float* y = (const float*)d_in[1];
  const float* g = (const float*)d_in[2];
  float* out = (float*)d_out;

  // Workspace layout: xh (1 MB) | yh (1 MB) | x2 (32 KB) | y2 (32 KB)
  _Float16* xh = (_Float16*)d_ws;
  _Float16* yh = xh + (size_t)N * D;
  float*    x2 = (float*)(yh + (size_t)M * D);
  float*    y2 = x2 + N;

  const int prepThreads = (N + M) * 32;                  // one wave per row
  rbf_prep<<<dim3(prepThreads / 256), dim3(256), 0, stream>>>(x, y, xh, yh, x2, y2);

  rbf_wmma<<<dim3(N / 128, M / 128), dim3(256), 0, stream>>>(xh, yh, x2, y2, g, out);
}
